// ResidualVectorQuantize_83124797047179
// MI455X (gfx1250) — matrix operand order repaired and checked by reading, not compiled, not run on hardware
//
#include <hip/hip_runtime.h>
#include <hip/hip_bf16.h>
#include <stdint.h>

// ---------------- problem constants ----------------
#define NUM_Q 8
#define KCB   1024          // codebook size
#define DDIM  512           // feature dim
#define BT    32000         // B*T tokens
#define BETA  0.25f

// ---------------- WMMA types ----------------
typedef __bf16  bf16x4  __attribute__((ext_vector_type(4)));
typedef __bf16  bf16x8  __attribute__((ext_vector_type(8)));
typedef __bf16  bf16x16 __attribute__((ext_vector_type(16)));
typedef float   f32x8   __attribute__((ext_vector_type(8)));

static __device__ __forceinline__ bf16x16 cat8(bf16x8 a, bf16x8 b) {
    bf16x16 r;
#pragma unroll
    for (int i = 0; i < 8; ++i) { r[i] = a[i]; r[i + 8] = b[i]; }
    return r;
}

#define WMMA_BF16(A, B, C)                                                   \
    __builtin_amdgcn_wmma_f32_16x16x32_bf16(false, (A), false, (B),          \
                                            (short)0, (C), false, false)

// =====================================================================
// init: residual = x ; q_out = 0 ; loss accumulators = 0
// grid: 16000 x 256, 4 floats/thread (16,384,000 elements exactly)
// =====================================================================
__global__ void rvq_init(const float* __restrict__ x,
                         float* __restrict__ resid,
                         float* __restrict__ qout,
                         float* __restrict__ lossAcc) {
    size_t gid  = (size_t)blockIdx.x * blockDim.x + threadIdx.x;
    size_t base = gid * 4;
    float4 v = *(const float4*)(x + base);
    *(float4*)(resid + base) = v;
    float4 z = make_float4(0.f, 0.f, 0.f, 0.f);
    *(float4*)(qout + base) = z;
    if (blockIdx.x == 0 && threadIdx.x < NUM_Q) lossAcc[threadIdx.x] = 0.f;
}

// =====================================================================
// prep_e2: ||e||^2 per code row, one wave per row (8192 rows)
// grid: 1024 x 256 (8 waves/block)
// =====================================================================
__global__ void rvq_prep_e2(const float* __restrict__ cb,
                            float* __restrict__ e2) {
    int wave = threadIdx.x >> 5;
    int lane = threadIdx.x & 31;
    int row  = blockIdx.x * 8 + wave;            // [0, NUM_Q*KCB)
    const float* p = cb + (size_t)row * DDIM;
    float s = 0.f;
#pragma unroll 4
    for (int j = lane; j < DDIM; j += 32) { float v = p[j]; s += v * v; }
#pragma unroll
    for (int m = 16; m >= 1; m >>= 1) s += __shfl_xor(s, m);
    if (lane == 0) e2[row] = s;
}

// =====================================================================
// prep_cb: split fp32 codebooks into bf16 hi + bf16 lo
// grid: 4096 x 256, 4 elements/thread (4,194,304 elements exactly)
// =====================================================================
__global__ void rvq_prep_cb(const float* __restrict__ cb,
                            __bf16* __restrict__ cbh,
                            __bf16* __restrict__ cbl) {
    size_t gid  = (size_t)blockIdx.x * blockDim.x + threadIdx.x;
    size_t base = gid * 4;
    float4 v = *(const float4*)(cb + base);
    float vv[4] = {v.x, v.y, v.z, v.w};
    bf16x4 h, l;
#pragma unroll
    for (int j = 0; j < 4; ++j) {
        __bf16 hj = (__bf16)vv[j];
        h[j] = hj;
        l[j] = (__bf16)(vv[j] - (float)hj);
    }
    *(bf16x4*)(cbh + base) = h;
    *(bf16x4*)(cbl + base) = l;
}

// =====================================================================
// argmin: per stage, nearest-code search via bf16 hi/lo WMMA
//   2x2 register blocking per wave: 32 M-rows x 32 N-codes per step
//   block = 64 (2 waves, 32 rows each), grid = 500 (500*64 == 32000)
//   dyn LDS = 2 * 64 * 520 * sizeof(bf16) = 133,120 B
// =====================================================================
#define M_TILE  64
#define LDS_STR 520   // 512 + 8 halfs pad -> conflict-free ds_load_b128

__global__ void __launch_bounds__(64)
rvq_argmin(const float* __restrict__ resid,
           const __bf16* __restrict__ cbh,
           const __bf16* __restrict__ cbl,
           const float* __restrict__ e2,
           int* __restrict__ idxOut,
           int s) {
    extern __shared__ __bf16 smem[];
    __bf16* Ah = smem;
    __bf16* Al = smem + M_TILE * LDS_STR;

    const int tid  = threadIdx.x;
    const int tok0 = blockIdx.x * M_TILE;

    // stage residual tile into LDS as bf16 hi/lo
    for (int i = tid; i < M_TILE * DDIM; i += blockDim.x) {
        int r = i >> 9;          // / 512
        int d = i & (DDIM - 1);
        float v  = resid[(size_t)(tok0 + r) * DDIM + d];
        __bf16 h = (__bf16)v;
        Ah[r * LDS_STR + d] = h;
        Al[r * LDS_STR + d] = (__bf16)(v - (float)h);
    }
    __syncthreads();

    const int wave = tid >> 5;
    const int lane = tid & 31;
    const int ln   = lane & 15;   // column / A-row index within a 16-tile
    const int hl   = lane >> 4;   // K-half select

    // wave owns rows [wave*32, wave*32+32): two 16-row A tiles
    const int row0 = wave * 32 + ln;       // A tile 0
    const int row1 = wave * 32 + 16 + ln;  // A tile 1
    const __bf16* a0h = Ah + row0 * LDS_STR;
    const __bf16* a0l = Al + row0 * LDS_STR;
    const __bf16* a1h = Ah + row1 * LDS_STR;
    const __bf16* a1l = Al + row1 * LDS_STR;

    float minv0[8], minv1[8];
    int   mini0[8], mini1[8];
#pragma unroll
    for (int r = 0; r < 8; ++r) {
        minv0[r] = 3.4e38f; mini0[r] = 0;
        minv1[r] = 3.4e38f; mini1[r] = 0;
    }

    const size_t cbStage = (size_t)s * KCB * DDIM;
    const float* e2Stage = e2 + s * KCB;

#pragma unroll 1
    for (int nt = 0; nt < KCB / 32; ++nt) {
        const int code0 = nt * 32 + ln;        // B tile 0
        const int code1 = nt * 32 + 16 + ln;   // B tile 1
        // B fragment: lane = column (code); lanes 0-15 K={0..7,16..23}, lanes 16-31 K={8..15,24..31}
        const __bf16* b0hp = cbh + cbStage + (size_t)code0 * DDIM + hl * 8;
        const __bf16* b0lp = cbl + cbStage + (size_t)code0 * DDIM + hl * 8;
        const __bf16* b1hp = cbh + cbStage + (size_t)code1 * DDIM + hl * 8;
        const __bf16* b1lp = cbl + cbStage + (size_t)code1 * DDIM + hl * 8;

        f32x8 c00 = {}, c01 = {}, c10 = {}, c11 = {};
        // unroll 2: software-pipeline register sets so the scheduler can
        // overwrite A/B fragment VGPRs from the *other* iteration, hiding the
        // 4-coexec WMMA->VALU WAR hazard (removes v_nop padding).
#pragma unroll 2
        for (int dk = 0; dk < DDIM / 32; ++dk) {
            // A fragments per ISA 16-bit 16x32 layout: lanes 0-15 K={0..7,16..23},
            // lanes 16-31 K={8..15,24..31}
            bf16x16 fa0h = cat8(*(const bf16x8*)(a0h + dk * 32 + hl * 8),
                                *(const bf16x8*)(a0h + dk * 32 + 16 + hl * 8));
            bf16x16 fa0l = cat8(*(const bf16x8*)(a0l + dk * 32 + hl * 8),
                                *(const bf16x8*)(a0l + dk * 32 + 16 + hl * 8));
            bf16x16 fa1h = cat8(*(const bf16x8*)(a1h + dk * 32 + hl * 8),
                                *(const bf16x8*)(a1h + dk * 32 + 16 + hl * 8));
            bf16x16 fa1l = cat8(*(const bf16x8*)(a1l + dk * 32 + hl * 8),
                                *(const bf16x8*)(a1l + dk * 32 + 16 + hl * 8));
            bf16x16 fb0h = cat8(*(const bf16x8*)(b0hp + dk * 32), *(const bf16x8*)(b0hp + dk * 32 + 16));
            bf16x16 fb0l = cat8(*(const bf16x8*)(b0lp + dk * 32), *(const bf16x8*)(b0lp + dk * 32 + 16));
            bf16x16 fb1h = cat8(*(const bf16x8*)(b1hp + dk * 32), *(const bf16x8*)(b1hp + dk * 32 + 16));
            bf16x16 fb1l = cat8(*(const bf16x8*)(b1lp + dk * 32), *(const bf16x8*)(b1lp + dk * 32 + 16));
            // dot = hi*hi + hi*lo + lo*hi  (ll term negligible); 12 WMMAs / 16 loads
            c00 = WMMA_BF16(fa0h, fb0h, c00);
            c01 = WMMA_BF16(fa0h, fb1h, c01);
            c10 = WMMA_BF16(fa1h, fb0h, c10);
            c11 = WMMA_BF16(fa1h, fb1h, c11);
            c00 = WMMA_BF16(fa0h, fb0l, c00);
            c01 = WMMA_BF16(fa0h, fb1l, c01);
            c10 = WMMA_BF16(fa1h, fb0l, c10);
            c11 = WMMA_BF16(fa1h, fb1l, c11);
            c00 = WMMA_BF16(fa0l, fb0h, c00);
            c01 = WMMA_BF16(fa0l, fb1h, c01);
            c10 = WMMA_BF16(fa1l, fb0h, c10);
            c11 = WMMA_BF16(fa1l, fb1h, c11);
        }
        // dist(m, code) = ||e||^2 - 2 * dot   (||r||^2 constant per row)
        float e0 = e2Stage[code0];
        float e1 = e2Stage[code1];
#pragma unroll
        for (int r = 0; r < 8; ++r) {
            float d00 = __builtin_fmaf(-2.0f, c00[r], e0);
            float d01 = __builtin_fmaf(-2.0f, c01[r], e1);
            float d10 = __builtin_fmaf(-2.0f, c10[r], e0);
            float d11 = __builtin_fmaf(-2.0f, c11[r], e1);
            if (d00 < minv0[r]) { minv0[r] = d00; mini0[r] = code0; }
            if (d01 < minv0[r]) { minv0[r] = d01; mini0[r] = code1; }
            if (d10 < minv1[r]) { minv1[r] = d10; mini1[r] = code0; }
            if (d11 < minv1[r]) { minv1[r] = d11; mini1[r] = code1; }
        }
    }

    // reduce (min, argmin) across the 16 lanes of each half-wave;
    // masks <= 8 keep the two 16-lane groups (different rows) disjoint
#pragma unroll
    for (int r = 0; r < 8; ++r) {
#pragma unroll
        for (int m = 8; m >= 1; m >>= 1) {
            float ov0 = __shfl_xor(minv0[r], m);
            int   oi0 = __shfl_xor(mini0[r], m);
            if (ov0 < minv0[r] || (ov0 == minv0[r] && oi0 < mini0[r])) {
                minv0[r] = ov0; mini0[r] = oi0;
            }
            float ov1 = __shfl_xor(minv1[r], m);
            int   oi1 = __shfl_xor(mini1[r], m);
            if (ov1 < minv1[r] || (ov1 == minv1[r] && oi1 < mini1[r])) {
                minv1[r] = ov1; mini1[r] = oi1;
            }
        }
    }
    if (ln == 0) {
        // C layout: VGPR r -> row r (lanes 0-15) / row 8+r (lanes 16-31)
#pragma unroll
        for (int r = 0; r < 8; ++r) {
            int tokA = tok0 + wave * 32 + hl * 8 + r;        // A tile 0 rows
            int tokB = tok0 + wave * 32 + 16 + hl * 8 + r;   // A tile 1 rows
            idxOut[s * BT + tokA] = mini0[r];
            idxOut[s * BT + tokB] = mini1[r];
        }
    }
}

// =====================================================================
// update: gather codeword, residual -= q, q_out += q, loss += |q - r|^2,
// emit indices-as-float. grid: 16000 x 256, 4 floats/thread.
// =====================================================================
__global__ void rvq_update(const float* __restrict__ cb,
                           const int* __restrict__ idx,
                           float* __restrict__ resid,
                           float* __restrict__ qout,
                           float* __restrict__ outIdx,
                           float* __restrict__ lossAcc,
                           int s) {
    size_t gid  = (size_t)blockIdx.x * blockDim.x + threadIdx.x;
    size_t base = gid * 4;
    int tok = (int)(base >> 9);       // / 512
    int d   = (int)(base & (DDIM - 1));

    int k = idx[s * BT + tok];
    float4 q = *(const float4*)(cb + ((size_t)s * KCB + k) * DDIM + d);
    float4 r = *(const float4*)(resid + base);

    float dx = q.x - r.x, dy = q.y - r.y, dz = q.z - r.z, dw = q.w - r.w;
    float ls = dx * dx + dy * dy + dz * dz + dw * dw;

    float4 rn = make_float4(r.x - q.x, r.y - q.y, r.z - q.z, r.w - q.w);
    *(float4*)(resid + base) = rn;

    float4 o = *(float4*)(qout + base);
    o.x += q.x; o.y += q.y; o.z += q.z; o.w += q.w;
    *(float4*)(qout + base) = o;

    if (d == 0) outIdx[s * BT + tok] = (float)k;

#pragma unroll
    for (int m = 16; m >= 1; m >>= 1) ls += __shfl_xor(ls, m);
    if ((threadIdx.x & 31) == 0) atomicAdd(lossAcc + s, ls);
}

// =====================================================================
// finalize: mean loss = (1+beta) * sum / (NUM_Q * B*T*D)
// =====================================================================
__global__ void rvq_finalize(const float* __restrict__ lossAcc,
                             float* __restrict__ outLoss) {
    if (threadIdx.x == 0) {
        float t = 0.f;
#pragma unroll
        for (int i = 0; i < NUM_Q; ++i) t += lossAcc[i];
        outLoss[0] = t * (1.0f + BETA) / ((float)NUM_Q * (float)BT * (float)DDIM);
    }
}

// =====================================================================
extern "C" void kernel_launch(void* const* d_in, const int* in_sizes, int n_in,
                              void* d_out, int out_size, void* d_ws, size_t ws_size,
                              hipStream_t stream) {
    (void)in_sizes; (void)n_in; (void)out_size; (void)ws_size;

    const float* x  = (const float*)d_in[0];                    // [B,T,D]
    const float* cb = (const float*)d_in[1];                    // [NUM_Q,K,D]

    // ---- workspace layout (all offsets 32B-aligned) ----
    char* ws = (char*)d_ws;
    float*  resid = (float*)ws;                                  // 65,536,000 B
    __bf16* cbh   = (__bf16*)(ws + 65536000u);                   //  8,388,608 B
    __bf16* cbl   = (__bf16*)(ws + 65536000u + 8388608u);        //  8,388,608 B
    float*  e2    = (float*) (ws + 65536000u + 2u * 8388608u);   //     32,768 B
    int*    idx   = (int*)   (ws + 65536000u + 2u * 8388608u + 32768u);   // 1,024,000 B
    float*  loss  = (float*) (ws + 65536000u + 2u * 8388608u + 32768u + 1024000u);

    // ---- output layout: q_out | indices (as float) | loss ----
    float* qout    = (float*)d_out;              // 16,384,000
    float* outIdx  = qout + (size_t)BT * DDIM;   // 256,000
    float* outLoss = outIdx + (size_t)NUM_Q * BT;

    rvq_init<<<16000, 256, 0, stream>>>(x, resid, qout, loss);
    rvq_prep_e2<<<1024, 256, 0, stream>>>(cb, e2);
    rvq_prep_cb<<<4096, 256, 0, stream>>>(cb, cbh, cbl);

    const size_t ldsBytes = 2u * M_TILE * LDS_STR * sizeof(__bf16); // 133,120 B
    for (int s = 0; s < NUM_Q; ++s) {
        rvq_argmin<<<BT / M_TILE, 64, ldsBytes, stream>>>(resid, cbh, cbl, e2, idx, s);
        rvq_update<<<16000, 256, 0, stream>>>(cb, idx, resid, qout, outIdx, loss, s);
    }
    rvq_finalize<<<1, 32, 0, stream>>>(loss, outLoss);
}
